// CausalAttention_40192303956227
// MI455X (gfx1250) — matrix operand order repaired and checked by reading, not compiled, not run on hardware
//
#include <hip/hip_runtime.h>

#define DEVINL __device__ __forceinline__

typedef __attribute__((ext_vector_type(16))) __bf16 v16bf;
typedef __attribute__((ext_vector_type(8)))  float  v8f;

union FragU { v16bf v; unsigned int d[8]; };

DEVINL unsigned short f2bf(float f) {
  unsigned int u = __float_as_uint(f);
  u += 0x7FFFu + ((u >> 16) & 1u);   // round-to-nearest-even
  return (unsigned short)(u >> 16);
}

DEVINL v8f vzero() {
  v8f z;
#pragma unroll
  for (int i = 0; i < 8; ++i) z[i] = 0.0f;
  return z;
}

// Low 32 bits of a generic LDS pointer == LDS byte offset (aperture only
// occupies addr[63:32] on gfx1250).
DEVINL unsigned lds_off(const void* p) { return (unsigned)(size_t)p; }

// CDNA5 async global->LDS (tracked by ASYNCcnt). vdst = LDS byte address.
DEVINL void async_load_b128(unsigned lds, const void* gaddr) {
  asm volatile("global_load_async_to_lds_b128 %0, %1, off"
               :: "v"(lds), "v"((unsigned long long)(size_t)gaddr)
               : "memory");
}
DEVINL void wait_async0() {
  asm volatile("s_wait_asynccnt 0" ::: "memory");
}

// A-matrix fragment (16x32 bf16, row-major source, stride ld in halves).
// Lane l: m = l%16, half = l/16. VGPR p holds K = 16*(p>>2) + 8*half + 2*(p&3).
DEVINL v16bf load_frag_a(const unsigned short* base, int ld) {
  const int lane = threadIdx.x & 31;
  const int m = lane & 15, h = lane >> 4;
  const unsigned short* row = base + m * ld;
  FragU f;
#pragma unroll
  for (int p = 0; p < 8; ++p) {
    const int k = ((p >> 2) << 4) + (h << 3) + ((p & 3) << 1);
    f.d[p] = *(const unsigned int*)(row + k);
  }
  return f.v;
}

// B-matrix fragment (32x16 bf16). Source row n holds the K-run for output
// column n. Lane l: n = l%16, half = l/16. VGPR p holds K = 16*(p>>2) + 8*half + 2*(p&3).
DEVINL v16bf load_frag_b(const unsigned short* base, int ld) {
  const int lane = threadIdx.x & 31;
  const int n = lane & 15, h = lane >> 4;
  const unsigned short* row = base + n * ld;
  FragU f;
#pragma unroll
  for (int p = 0; p < 8; ++p) {
    const int k = ((p >> 2) << 4) + (h << 3) + ((p & 3) << 1);
    f.d[p] = *(const unsigned int*)(row + k);
  }
  return f.v;
}

DEVINL v8f wmma_bf16(v16bf a, v16bf b, v8f c) {
  return __builtin_amdgcn_wmma_f32_16x16x32_bf16(false, a, false, b, (short)0, c,
                                                 false, false);
}

DEVINL float rowmax16(float v) {
#pragma unroll
  for (int m = 8; m >= 1; m >>= 1) v = fmaxf(v, __shfl_xor(v, m, 32));
  return v;
}
DEVINL float rowsum16(float v) {
#pragma unroll
  for (int m = 8; m >= 1; m >>= 1) v += __shfl_xor(v, m, 32);
  return v;
}

// ---------------------------------------------------------------- convert
// 8 floats -> 8 bf16 per thread (float4 x2 in, uint4 out).
__global__ void cvt_bf16_kernel(const float* __restrict__ src,
                                unsigned short* __restrict__ dst, int n8) {
  const int i = blockIdx.x * blockDim.x + threadIdx.x;
  if (i < n8) {
    const float4* s = (const float4*)(src + (size_t)i * 8);
    const float4 a = s[0], b = s[1];
    union { uint4 q; unsigned short h[8]; } u;
    u.h[0] = f2bf(a.x); u.h[1] = f2bf(a.y); u.h[2] = f2bf(a.z); u.h[3] = f2bf(a.w);
    u.h[4] = f2bf(b.x); u.h[5] = f2bf(b.y); u.h[6] = f2bf(b.z); u.h[7] = f2bf(b.w);
    *(uint4*)(dst + (size_t)i * 8) = u.q;
  }
}

// ---------------------------------------------------------------- GEMM
// C[128 x 64] per block; 8 waves, wave w owns rows [w*16, w*16+16).
// Double-buffered K-loop: A tile via async global->LDS, B tile transposed
// through VGPRs. MODE 0: f32 row-major out. MODE 1: QKV scatter epilogue.
template <int MODE>
__global__ __launch_bounds__(256) void gemm128x64(
    const unsigned short* __restrict__ A, const unsigned short* __restrict__ Bw,
    int N, int K, float* __restrict__ outf, unsigned short* __restrict__ qb,
    unsigned short* __restrict__ kbuf, unsigned short* __restrict__ vtb) {
  const int LDA = 40, LDB = 40;
  __shared__ unsigned short s_a[2][128 * LDA];  // A tile 128x32, row-major
  __shared__ unsigned short s_b[2][64 * LDB];   // B tile transposed: [n][k]

  const int tid = threadIdx.x;
  const int w = tid >> 5, lane = tid & 31;
  const int nl = lane & 15, half = lane >> 4;
  const int m0 = blockIdx.y * 128;
  const int n0 = blockIdx.x * 64;

  v8f acc[4];
#pragma unroll
  for (int j = 0; j < 4; ++j) acc[j] = vzero();

  // Stage helpers (uniform across the block).
  auto stage_a = [&](int k0, unsigned short* sa) {
#pragma unroll
    for (int i = 0; i < 2; ++i) {
      const int idx = (tid << 1) + i;
      const int r = idx >> 2, kq = (idx & 3) << 3;
      async_load_b128(lds_off(&sa[r * LDA + kq]),
                      A + (size_t)(m0 + r) * K + k0 + kq);
    }
  };
  auto stage_b = [&](int k0, unsigned short* sb) {
    const int kr = tid >> 3, nq = (tid & 7) << 3;
    union { uint4 q; unsigned short h[8]; } u;
    u.q = *(const uint4*)(Bw + (size_t)(k0 + kr) * N + n0 + nq);
#pragma unroll
    for (int e = 0; e < 8; ++e) sb[(nq + e) * LDB + kr] = u.h[e];
  };

  stage_a(0, s_a[0]);
  stage_b(0, s_b[0]);

  int buf = 0;
  for (int k0 = 0; k0 < K; k0 += 32, buf ^= 1) {
    wait_async0();       // own async ops for tile k0 complete
    __syncthreads();     // everyone's (also orders the ds stores of stage_b)
    if (k0 + 32 < K) {   // prefetch next tile into the other buffer
      stage_a(k0 + 32, s_a[buf ^ 1]);
      stage_b(k0 + 32, s_b[buf ^ 1]);
    }
    const v16bf af = load_frag_a(&s_a[buf][(w * 16) * LDA], LDA);
#pragma unroll
    for (int j = 0; j < 4; ++j) {
      const v16bf bf = load_frag_b(&s_b[buf][(j * 16) * LDB], LDB);
      acc[j] = wmma_bf16(af, bf, acc[j]);
    }
  }

#pragma unroll
  for (int j = 0; j < 4; ++j) {
#pragma unroll
    for (int r = 0; r < 8; ++r) {
      const int m_g = m0 + w * 16 + r + (half << 3);
      const int c = n0 + j * 16 + nl;
      const float v = acc[j][r];
      if (MODE == 0) {
        outf[(size_t)m_g * N + c] = v;
      } else {
        const int b = m_g >> 11, t = m_g & 2047;   // T = 2048
        const int which = c >> 10, cc = c & 1023;  // D = 1024
        const int h = cc >> 6, d = cc & 63;        // HD = 64
        const int bh = b * 16 + h;                 // H = 16
        if (which == 0)
          qb[((size_t)bh * 2048 + t) * 64 + d] = f2bf(v * 0.125f);
        else if (which == 1)
          kbuf[((size_t)bh * 2048 + t) * 64 + d] = f2bf(v);
        else
          vtb[((size_t)bh * 64 + d) * 2048 + t] = f2bf(v);
      }
    }
  }
}

// ---------------------------------------------------------------- attention
// Block: one (b,h), 128 query rows (8 waves x 16). 32-key blocks, K/Vt tiles
// double-buffered via async global->LDS.
__global__ __launch_bounds__(256) void attn_fa(
    const unsigned short* __restrict__ qg, const unsigned short* __restrict__ kg,
    const unsigned short* __restrict__ vtg, unsigned short* __restrict__ og) {
  const int T = 2048, Dm = 1024, Hh = 16;
  const int LDK = 72, LDV = 40, LDP = 40;
  __shared__ unsigned short s_k[2][32 * LDK];  // K block: [key][d]
  __shared__ unsigned short s_v[2][64 * LDV];  // Vt block: [d][key]
  __shared__ unsigned short s_p[8][16 * LDP];  // per-wave P tile: [m][key]

  const int bh = blockIdx.y;
  const int qblk = blockIdx.x;
  const int tid = threadIdx.x;
  const int w = tid >> 5, lane = tid & 31;
  const int nl = lane & 15, half = lane >> 4;
  const int qbase = qblk * 128 + w * 16;

  // Q fragments (pre-scaled by 1/sqrt(HD) in the QKV epilogue).
  const unsigned short* qrow = qg + ((size_t)bh * T + qbase) * 64;
  const v16bf q0 = load_frag_a(qrow, 64);
  const v16bf q1 = load_frag_a(qrow + 32, 64);

  const float NEG = -1.0e30f;
  float mrow[8], lrow[8];
  v8f o[4];
#pragma unroll
  for (int r = 0; r < 8; ++r) { mrow[r] = NEG; lrow[r] = 0.0f; }
#pragma unroll
  for (int j = 0; j < 4; ++j) o[j] = vzero();

  auto stage_kv = [&](int kb, unsigned short* sk, unsigned short* sv) {
    const int kr = tid >> 3, dq = (tid & 7) << 3;
    async_load_b128(lds_off(&sk[kr * LDK + dq]),
                    kg + ((size_t)bh * T + kb + kr) * 64 + dq);
    const int dv = tid >> 2, tq = (tid & 3) << 3;
    async_load_b128(lds_off(&sv[dv * LDV + tq]),
                    vtg + ((size_t)bh * 64 + dv) * T + kb + tq);
  };

  const int kb_end = qblk * 128 + 128;
  stage_kv(0, s_k[0], s_v[0]);

  int buf = 0;
  for (int kb = 0; kb < kb_end; kb += 32, buf ^= 1) {
    wait_async0();
    __syncthreads();
    if (kb + 32 < kb_end) stage_kv(kb + 32, s_k[buf ^ 1], s_v[buf ^ 1]);

    if (kb <= qbase + 15) {  // causally needed by this wave
      // S = Q @ K^T : two 16-key tiles, contraction over HD=64 in two chunks.
      v8f c0 = vzero(), c1 = vzero();
      {
        const unsigned short* sk = s_k[buf];
        const v16bf b00 = load_frag_b(&sk[0], LDK);
        const v16bf b01 = load_frag_b(&sk[32], LDK);
        c0 = wmma_bf16(q0, b00, c0);
        c0 = wmma_bf16(q1, b01, c0);
        const v16bf b10 = load_frag_b(&sk[16 * LDK], LDK);
        const v16bf b11 = load_frag_b(&sk[16 * LDK + 32], LDK);
        c1 = wmma_bf16(q0, b10, c1);
        c1 = wmma_bf16(q1, b11, c1);
      }
      if (kb + 31 > qbase) {  // partially masked block
#pragma unroll
        for (int r = 0; r < 8; ++r) {
          const int q = qbase + r + (half << 3);
          if (kb + nl > q) c0[r] = NEG;
          if (kb + 16 + nl > q) c1[r] = NEG;
        }
      }
      // Online softmax per row; build bf16 P tile in LDS.
#pragma unroll
      for (int r = 0; r < 8; ++r) {
        const float s0 = c0[r], s1 = c1[r];
        const float bm = rowmax16(fmaxf(s0, s1));
        const float mnew = fmaxf(mrow[r], bm);
        const float alpha = __expf(mrow[r] - mnew);
        const float p0 = __expf(s0 - mnew);
        const float p1 = __expf(s1 - mnew);
        const float ps = rowsum16(p0 + p1);
        lrow[r] = lrow[r] * alpha + ps;
        mrow[r] = mnew;
#pragma unroll
        for (int j = 0; j < 4; ++j) o[j][r] *= alpha;
        unsigned short* pr = &s_p[w][(r + (half << 3)) * LDP];
        pr[nl] = f2bf(p0);
        pr[16 + nl] = f2bf(p1);
      }
      asm volatile("s_wait_dscnt 0" ::: "memory");  // P visible to own wave
      // O += P @ V : one A-fragment (16x32 keys) x four dim-group B-fragments.
      const v16bf pf = load_frag_a(&s_p[w][0], LDP);
#pragma unroll
      for (int j = 0; j < 4; ++j) {
        const v16bf vf = load_frag_b(&s_v[buf][(j * 16) * LDV], LDV);
        o[j] = wmma_bf16(pf, vf, o[j]);
      }
    }
  }

  // Normalize and write bf16 attention output in [B, T, D] layout.
  float invl[8];
#pragma unroll
  for (int r = 0; r < 8; ++r) invl[r] = 1.0f / lrow[r];
  const int b = bh / Hh, h = bh % Hh;
#pragma unroll
  for (int j = 0; j < 4; ++j) {
#pragma unroll
    for (int r = 0; r < 8; ++r) {
      const int qrowg = qbase + r + (half << 3);
      const int d = j * 16 + nl;
      og[((size_t)(b * T + qrowg)) * Dm + h * 64 + d] = f2bf(o[j][r] * invl[r]);
    }
  }
}

// ---------------------------------------------------------------- launch
extern "C" void kernel_launch(void* const* d_in, const int* in_sizes, int n_in,
                              void* d_out, int out_size, void* d_ws,
                              size_t ws_size, hipStream_t stream) {
  (void)in_sizes; (void)n_in; (void)out_size; (void)ws_size;
  const float* x = (const float*)d_in[0];
  const float* wqkv = (const float*)d_in[1];
  const float* wout = (const float*)d_in[2];
  float* out = (float*)d_out;

  const int Bt = 4, T = 2048, D = 1024, H = 16;
  const size_t nX = (size_t)Bt * T * D;   // 8388608
  const size_t nWq = (size_t)D * 3 * D;   // 3145728
  const size_t nWo = (size_t)D * D;       // 1048576

  unsigned short* xb = (unsigned short*)d_ws;
  unsigned short* wqb = xb + nX;
  unsigned short* wob = wqb + nWq;
  unsigned short* qb = wob + nWo;
  unsigned short* kbuf = qb + nX;
  unsigned short* vtb = kbuf + nX;
  unsigned short* ab = vtb + nX;  // total ~92 MB of workspace

  cvt_bf16_kernel<<<(int)(nX / 8 / 256), 256, 0, stream>>>(x, xb, (int)(nX / 8));
  cvt_bf16_kernel<<<(int)(nWq / 8 / 256), 256, 0, stream>>>(wqkv, wqb,
                                                            (int)(nWq / 8));
  cvt_bf16_kernel<<<(int)(nWo / 8 / 256), 256, 0, stream>>>(wout, wob,
                                                            (int)(nWo / 8));

  dim3 g1(3 * D / 64, (Bt * T) / 128);
  gemm128x64<1><<<g1, 256, 0, stream>>>(xb, wqb, 3 * D, D, nullptr, qb, kbuf, vtb);

  dim3 ga(T / 128, Bt * H);
  attn_fa<<<ga, 256, 0, stream>>>(qb, kbuf, vtb, ab);

  dim3 g2(D / 64, (Bt * T) / 128);
  gemm128x64<0><<<g2, 256, 0, stream>>>(ab, wob, D, D, out, nullptr, nullptr,
                                        nullptr);
}
